// InstanceHead_7473243095301
// MI455X (gfx1250) — matrix operand order repaired and checked by reading, not compiled, not run on hardware
//
#include <hip/hip_runtime.h>

// ---------------------------------------------------------------------------
// InstanceHead on MI455X (gfx1250, wave32).
//
// Algebraic restructuring:  pre[n,m,:] = P[n,:] + Q[m,:]   (dists term is
// separable), so the O(N*M) sweep only does relu(P+Q)·W2 per pair.
//
// Roofline: 23 MB output @ 23.3 TB/s => ~1.2 us floor; ~0.35 GFLOP pairwise
// => write-bandwidth bound.  P-precompute is a real GEMM (60000x22 @ 22x19)
// and is done with v_wmma_f32_16x16x32_f16 (K=22 zero-padded to 32, f32
// accumulate), two WMMAs per 16-row tile.
// ---------------------------------------------------------------------------

typedef __attribute__((ext_vector_type(16))) _Float16 v16h;
typedef __attribute__((ext_vector_type(8)))  float    v8f;

#define N_VOX  60000
#define M_CENT 96
#define D_DESC 16
#define C_IN   39     // 2*D + 7
#define C_HID  19     // D + 3
#define HPAD   20     // padded hidden stride (80 B = 5 x float4)

// ---------------------------------------------------------------------------
// Kernel A: Q[m][h] = cent_desc[m]·W1c[h] - cent_coords[m]·W1d[h] + b1[h]
//           (Q[m][19] = 0 pad), plus cb[m] = batch_ids[peak[m]].
// Tiny (96x19); one block.
// ---------------------------------------------------------------------------
__global__ void prep_q_kernel(const float* __restrict__ voxel_desc,
                              const float* __restrict__ centroid_conf,
                              const float* __restrict__ W1,
                              const float* __restrict__ b1,
                              const int*   __restrict__ coords,
                              const int*   __restrict__ batch_ids,
                              const int*   __restrict__ peaks,
                              float* __restrict__ Q,
                              int*   __restrict__ cb)
{
    const int tid = threadIdx.x;
    for (int i = tid; i < M_CENT * HPAD; i += blockDim.x) {
        const int m = i / HPAD;
        const int h = i % HPAD;
        float q = 0.0f;
        if (h < C_HID) {
            const int p = peaks[m];
            const float* wrow = W1 + h * C_IN;
            float acc = b1[h];
            #pragma unroll
            for (int j = 0; j < D_DESC; ++j)                 // W1c cols 19..34
                acc += voxel_desc[p * D_DESC + j] * wrow[19 + j];
            acc += centroid_conf[m] * wrow[35];              // W1c col 35 (conf)
            #pragma unroll
            for (int k = 0; k < 3; ++k)                      // -cc·W1d
                acc -= (float)coords[p * 3 + k] * wrow[36 + k];
            q = acc;
        }
        Q[i] = q;
    }
    if (tid < M_CENT) cb[tid] = batch_ids[peaks[tid]];
}

// ---------------------------------------------------------------------------
// Kernel B: P = X @ Wp^T via v_wmma_f32_16x16x32_f16.
//   X(n, j):  j<3 offsets, j<19 voxel_desc, j<22 coords+offsets, else 0
//   Wp(h, k): k<19 W1[h][k] (W1v), k<22 W1[h][36+(k-19)] (W1d), else 0
// One 16-row tile per WMMA pair: c0 covers h 0..15, c1 covers h 16..31
// (h>=19 columns are zero => P[·][19] = 0 pad written for free).
// ---------------------------------------------------------------------------
__device__ __forceinline__ float xval(const float* __restrict__ off,
                                      const float* __restrict__ desc,
                                      const int*   __restrict__ coords,
                                      int n, int j)
{
    if (j < 3)  return off[n * 3 + j];
    if (j < 19) return desc[n * D_DESC + (j - 3)];
    if (j < 22) { const int k = j - 19; return (float)coords[n * 3 + k] + off[n * 3 + k]; }
    return 0.0f;
}

__device__ __forceinline__ float wval(const float* __restrict__ W1, int h, int k)
{
    if (h >= C_HID) return 0.0f;
    if (k < 19) return W1[h * C_IN + k];
    if (k < 22) return W1[h * C_IN + 36 + (k - 19)];
    return 0.0f;
}

__global__ void __launch_bounds__(128)
prep_p_kernel(const float* __restrict__ voxel_desc,
              const float* __restrict__ offsets,
              const float* __restrict__ W1,
              const int*   __restrict__ coords,
              float* __restrict__ P)
{
    const int lane   = threadIdx.x & 31;
    const int wave   = (blockIdx.x * blockDim.x + threadIdx.x) >> 5;
    const int nWaves = (gridDim.x * blockDim.x) >> 5;
    const int nTiles = N_VOX / 16;            // 3750, wave-uniform loop

    // B fragments (weights), built once per wave.
    // B 32x16 fragment, contiguous k order as used here (same as A): lane L -> N = L&15; half i -> K = (L<16?0:16)+i.
    const int nB = lane & 15;
    const int kB = (lane < 16) ? 0 : 16;
    v16h bf0, bf1;
    #pragma unroll
    for (int i = 0; i < 16; ++i) {
        bf0[i] = (_Float16)wval(W1, nB,      kB + i);   // h columns 0..15
        bf1[i] = (_Float16)wval(W1, nB + 16, kB + i);   // h columns 16..31 (19+ => 0)
    }

    // A 16x32 fragment, contiguous k order as used here (same as B): lane L -> M = L&15;
    // halves: L<16: K = 0..15;  L>=16: K = 16..31.
    const int kA0    = (lane < 16) ? 0 : 16;
    const int rowOff = (lane < 16) ? 0 : 8;   // C/D: lanes 16..31 hold M = r+8

    for (int t = wave; t < nTiles; t += nWaves) {
        const int n0 = t * 16;
        const int nA = n0 + (lane & 15);
        v16h a;
        #pragma unroll
        for (int i = 0; i < 16; ++i) {
            const int k = kA0 + i;
            a[i] = (_Float16)xval(offsets, voxel_desc, coords, nA, k);
        }
        v8f c0 = {}; v8f c1 = {};
        c0 = __builtin_amdgcn_wmma_f32_16x16x32_f16(false, a, false, bf0, (short)0, c0, false, false);
        c1 = __builtin_amdgcn_wmma_f32_16x16x32_f16(false, a, false, bf1, (short)0, c1, false, false);

        #pragma unroll
        for (int r = 0; r < 8; ++r) {
            const int row = n0 + r + rowOff;
            P[row * HPAD + nB] = c0[r];                       // h = 0..15
            if (nB < 4) P[row * HPAD + 16 + nB] = c1[r];      // h = 16..19 (19 -> 0)
        }
    }
}

// ---------------------------------------------------------------------------
// Kernel C: out[n][m] = mask( clip( b2 + sum_h relu(P[n,h]+Q[m,h]) * W2[h] ) )
// Block = (32, 8): 8 voxels x all 96 centroids (each lane does m, m+32, m+64).
// Q (7.7 KB) lives in LDS; P-row in registers via 5 aligned float4 loads.
// ---------------------------------------------------------------------------
__global__ void __launch_bounds__(256)
logits_kernel(const float* __restrict__ P,
              const float* __restrict__ Q,
              const int*   __restrict__ cb,
              const float* __restrict__ W2,
              const float* __restrict__ b2,
              const int*   __restrict__ batch_ids,
              float* __restrict__ out)
{
    __shared__ float Qs[M_CENT * HPAD];
    __shared__ float w2s[HPAD];
    __shared__ int   cbs[M_CENT];

    const int tx  = threadIdx.x;            // 0..31 : m lane
    const int ty  = threadIdx.y;            // 0..7  : n row
    const int tid = ty * 32 + tx;

    for (int i = tid; i < M_CENT * HPAD; i += 256) Qs[i] = Q[i];
    if (tid < M_CENT) cbs[tid] = cb[tid];
    if (tid < HPAD)   w2s[tid] = (tid < C_HID) ? W2[tid] : 0.0f;
    __syncthreads();

    const int   n   = blockIdx.x * 8 + ty;  // N_VOX = 7500 * 8 exactly
    const float b2v = b2[0];
    const int   bn  = batch_ids[n];

    float4 p[5];
    const float4* prow = reinterpret_cast<const float4*>(P + (size_t)n * HPAD);
    #pragma unroll
    for (int c = 0; c < 5; ++c) p[c] = prow[c];

    float w2r[HPAD];
    #pragma unroll
    for (int j = 0; j < HPAD; ++j) w2r[j] = w2s[j];

    #pragma unroll
    for (int mi = 0; mi < 3; ++mi) {
        const int m = tx + 32 * mi;
        const float4* qrow = reinterpret_cast<const float4*>(&Qs[m * HPAD]);
        float acc = b2v;
        #pragma unroll
        for (int c = 0; c < 5; ++c) {
            const float4 q = qrow[c];
            acc += fmaxf(p[c].x + q.x, 0.0f) * w2r[4 * c + 0];
            acc += fmaxf(p[c].y + q.y, 0.0f) * w2r[4 * c + 1];
            acc += fmaxf(p[c].z + q.z, 0.0f) * w2r[4 * c + 2];
            acc += fmaxf(p[c].w + q.w, 0.0f) * w2r[4 * c + 3];
        }
        float l = fminf(fmaxf(acc, -10.0f), 10.0f);
        l = (acc != acc) ? 0.0f : l;                       // nan_to_num(clip(x))
        l = (bn == cbs[m]) ? l : -__builtin_huge_valf();   // batch mask
        out[(size_t)n * M_CENT + m] = l;
    }
}

// ---------------------------------------------------------------------------
extern "C" void kernel_launch(void* const* d_in, const int* in_sizes, int n_in,
                              void* d_out, int out_size, void* d_ws, size_t ws_size,
                              hipStream_t stream)
{
    const float* voxel_desc    = (const float*)d_in[0];
    const float* centroid_conf = (const float*)d_in[1];
    const float* offsets       = (const float*)d_in[2];
    const float* W1            = (const float*)d_in[3];
    const float* b1            = (const float*)d_in[4];
    const float* W2            = (const float*)d_in[5];
    const float* b2            = (const float*)d_in[6];
    const int*   coords        = (const int*)d_in[7];
    const int*   batch_ids     = (const int*)d_in[8];
    const int*   peaks         = (const int*)d_in[9];
    float* out = (float*)d_out;

    // Workspace layout: P (60000*20 f32 = 4.8 MB) | Q (96*20 f32) | cb (96 i32)
    float* P  = (float*)d_ws;
    float* Q  = P + (size_t)N_VOX * HPAD;
    int*   cb = (int*)(Q + M_CENT * HPAD);

    prep_q_kernel<<<1, 256, 0, stream>>>(voxel_desc, centroid_conf, W1, b1,
                                         coords, batch_ids, peaks, Q, cb);
    prep_p_kernel<<<120, 128, 0, stream>>>(voxel_desc, offsets, W1, coords, P);
    logits_kernel<<<N_VOX / 8, dim3(32, 8), 0, stream>>>(P, Q, cb, W2, b2,
                                                         batch_ids, out);
}
